// VQ_89567247990968
// MI455X (gfx1250) — matrix operand order repaired and checked by reading, not compiled, not run on hardware
//
#include <hip/hip_runtime.h>

typedef __attribute__((ext_vector_type(16))) __bf16 v16bf;
typedef __attribute__((ext_vector_type(8)))  float  v8f;

#define K_CODES 1024
#define C_DIM   256
#define L_DIM   4096
#define N_TOK   65536            // 16 * 4096 tokens
#define TOK_PER_BLOCK 128
#define NBLK    (N_TOK / TOK_PER_BLOCK)   // 512 workgroups
#define KTILES  (K_CODES / 16)            // 64

// CDNA5 async global->LDS path (ASYNCcnt-tracked, bypasses VGPRs).
#if defined(__gfx1250__) && __has_builtin(__builtin_amdgcn_global_load_async_to_lds_b128)
#define HAVE_ASYNC_LDS 1
#else
#define HAVE_ASYNC_LDS 0
#endif

#if HAVE_ASYNC_LDS
#define AS_GLOBAL __attribute__((address_space(1)))
#define AS_LDS    __attribute__((address_space(3)))

__device__ __forceinline__ void async_cp16(const void* gsrc, void* ldst) {
  // Builtin prototype (from compiler diagnostic): param0 = int __vector(4) AS1*,
  // param1 = int __vector(4) AS3*, then imm offset, imm cpol.
  typedef int v4i __attribute__((vector_size(16)));
  __builtin_amdgcn_global_load_async_to_lds_b128(
      (AS_GLOBAL v4i*)(AS_GLOBAL const void*)gsrc,
      (AS_LDS    v4i*)(AS_LDS void*)ldst,
      0, 0);
}

__device__ __forceinline__ void async_wait_all() {
#if __has_builtin(__builtin_amdgcn_s_wait_asynccnt)
  __builtin_amdgcn_s_wait_asynccnt(0);
#else
  asm volatile("s_wait_asynccnt 0" ::: "memory");
#endif
}
#endif

// ---------------------------------------------------------------------------
// Prep: split codebook into bf16 hi/lo (e = hi + lo, RNE) and 0.5*||e||^2 f32.
// ---------------------------------------------------------------------------
__global__ void __launch_bounds__(C_DIM)
vq_prep(const float* __restrict__ cb,
        unsigned short* __restrict__ cb_hi_u,
        unsigned short* __restrict__ cb_lo_u,
        float* __restrict__ halfnorm) {
  const int k = blockIdx.x, c = threadIdx.x;
  float v = cb[(size_t)k * C_DIM + c];
  __bf16 h = (__bf16)v;
  float lo = v - (float)h;
  reinterpret_cast<__bf16*>(cb_hi_u)[(size_t)k * C_DIM + c] = h;
  reinterpret_cast<__bf16*>(cb_lo_u)[(size_t)k * C_DIM + c] = (__bf16)lo;
  __shared__ float red[C_DIM];
  red[c] = v * v;
  __syncthreads();
  for (int s = C_DIM / 2; s > 0; s >>= 1) {
    if (c < s) red[c] += red[c + s];
    __syncthreads();
  }
  if (c == 0) halfnorm[k] = 0.5f * red[0];
}

// ---------------------------------------------------------------------------
// Main: fused (z @ cb^T) score GEMM via bf16 hi/lo WMMA + running argmax,
// double-buffered B tiles in LDS (async global->LDS prefetch when available),
// then gather z_q -> out[B,C,L] and accumulate squared error per block.
// ---------------------------------------------------------------------------
__global__ void __launch_bounds__(256)
vq_main(const float* __restrict__ z,
        const float* __restrict__ cb,
        const unsigned short* __restrict__ cb_hi_u,
        const unsigned short* __restrict__ cb_lo_u,
        const float* __restrict__ halfnorm,
        float* __restrict__ out,
        float* __restrict__ partials) {
  __shared__ __align__(16) __bf16 lds_bhi[2][16 * C_DIM];  // 2 x 8 KB
  __shared__ __align__(16) __bf16 lds_blo[2][16 * C_DIM];  // 2 x 8 KB
  __shared__ int   lds_idx[TOK_PER_BLOCK];
  __shared__ float lds_red[256];

  const int tid  = threadIdx.x;
  const int wave = tid >> 5;
  const int lane = tid & 31;
  const int m    = lane & 15;     // A: token row within tile / B: code column
  const int half = lane >> 4;     // lane group (ISA WMMA layouts)
  const int tok0 = blockIdx.x * TOK_PER_BLOCK;

  // ---- Preload A fragments: 16 tokens x 256 channels, bf16 hi/lo ----------
  const int tokenA = tok0 + wave * 16 + m;
  const int bA = tokenA >> 12;
  const int lA = tokenA & (L_DIM - 1);
  const float* zg = z + (size_t)bA * (C_DIM * L_DIM) + lA;

  v16bf a_hi[8], a_lo[8];
#pragma unroll
  for (int cc = 0; cc < 8; ++cc) {
    v16bf hi, lo;
#pragma unroll
    for (int j = 0; j < 16; ++j) {
      // A k order used here (same as B): lane element j holds K = 16*half + j
      int c = cc * 32 + 16 * half + j;
      float zf = zg[(size_t)c * L_DIM];
      __bf16 h = (__bf16)zf;
      hi[j] = h;
      lo[j] = (__bf16)(zf - (float)h);
    }
    a_hi[cc] = hi;
    a_lo[cc] = lo;
  }

  float best[8];
  int   bidx[8];
#pragma unroll
  for (int r = 0; r < 8; ++r) { best[r] = -3.4e38f; bidx[r] = 0; }

  const uint4* ghi = (const uint4*)cb_hi_u;   // 512 uint4 per 16-code tile
  const uint4* glo = (const uint4*)cb_lo_u;

  // ---- prologue: stage tile 0 into buffer 0 -------------------------------
#if HAVE_ASYNC_LDS
  {
    uint4* dh = (uint4*)lds_bhi[0];
    uint4* dl = (uint4*)lds_blo[0];
    async_cp16(ghi + tid,       dh + tid);
    async_cp16(ghi + tid + 256, dh + tid + 256);
    async_cp16(glo + tid,       dl + tid);
    async_cp16(glo + tid + 256, dl + tid + 256);
    async_wait_all();
  }
#else
  {
    uint4* dh = (uint4*)lds_bhi[0];
    uint4* dl = (uint4*)lds_blo[0];
    dh[tid] = ghi[tid]; dh[tid + 256] = ghi[tid + 256];
    dl[tid] = glo[tid]; dl[tid + 256] = glo[tid + 256];
  }
#endif
  __syncthreads();

  // ---- K loop: 64 tiles of 16 codes, single barrier per iteration ---------
  for (int kt = 0; kt < KTILES; ++kt) {
    const int  buf  = kt & 1;
    const bool more = (kt + 1) < KTILES;
    const __bf16* bhi = lds_bhi[buf];
    const __bf16* blo = lds_blo[buf];

    // prefetch next B tile into the other buffer while we compute this one
#if HAVE_ASYNC_LDS
    if (more) {
      const uint4* sh = ghi + (size_t)(kt + 1) * 512;
      const uint4* sl = glo + (size_t)(kt + 1) * 512;
      uint4* dh = (uint4*)lds_bhi[buf ^ 1];
      uint4* dl = (uint4*)lds_blo[buf ^ 1];
      async_cp16(sh + tid,       dh + tid);
      async_cp16(sh + tid + 256, dh + tid + 256);
      async_cp16(sl + tid,       dl + tid);
      async_cp16(sl + tid + 256, dl + tid + 256);
    }
#else
    uint4 rh0 = {}, rh1 = {}, rl0 = {}, rl1 = {};
    if (more) {
      const uint4* sh = ghi + (size_t)(kt + 1) * 512;
      const uint4* sl = glo + (size_t)(kt + 1) * 512;
      rh0 = sh[tid]; rh1 = sh[tid + 256];
      rl0 = sl[tid]; rl1 = sl[tid + 256];
    }
#endif

    const float hn = halfnorm[kt * 16 + m];
    v8f acc = {};
#pragma unroll
    for (int cc = 0; cc < 8; ++cc) {
      // B layout: lane element j holds K-local = 16*half + j -> contiguous 32B
      union { v16bf v; uint4 u[2]; } bh, bl;
      const uint4* ph = (const uint4*)&bhi[m * C_DIM + cc * 32 + half * 16];
      const uint4* pl = (const uint4*)&blo[m * C_DIM + cc * 32 + half * 16];
      bh.u[0] = ph[0]; bh.u[1] = ph[1];
      bl.u[0] = pl[0]; bl.u[1] = pl[1];
      // fp32-accurate split product: zh*eh + zh*el + zl*eh (zl*el ~ 2^-18, dropped)
      acc = __builtin_amdgcn_wmma_f32_16x16x32_bf16(false, a_hi[cc], false, bh.v,
                                                    (short)0, acc, false, false);
      acc = __builtin_amdgcn_wmma_f32_16x16x32_bf16(false, a_hi[cc], false, bl.v,
                                                    (short)0, acc, false, false);
      acc = __builtin_amdgcn_wmma_f32_16x16x32_bf16(false, a_lo[cc], false, bh.v,
                                                    (short)0, acc, false, false);
    }

    const int cand = kt * 16 + m;   // code index this lane's column holds
#pragma unroll
    for (int r = 0; r < 8; ++r) {
      float s = acc[r] - hn;        // maximize z.e - 0.5||e||^2  == argmin dist
      if (s > best[r]) { best[r] = s; bidx[r] = cand; }  // strict: lowest k on tie
    }

#if HAVE_ASYNC_LDS
    if (more) async_wait_all();     // ASYNCcnt -> 0: next buffer resident in LDS
#else
    if (more) {
      uint4* dh = (uint4*)lds_bhi[buf ^ 1];
      uint4* dl = (uint4*)lds_blo[buf ^ 1];
      dh[tid] = rh0; dh[tid + 256] = rh1;
      dl[tid] = rl0; dl[tid + 256] = rl1;
    }
#endif
    __syncthreads();                // publish next buffer (single barrier/iter)
  }

  // ---- cross-lane argmax within each 16-lane group (C/D layout rows) ------
#pragma unroll
  for (int r = 0; r < 8; ++r) {
    float v = best[r];
    int  ix = bidx[r];
#pragma unroll
    for (int off = 8; off >= 1; off >>= 1) {
      float ov = __shfl_xor(v, off, 32);
      int  oix = __shfl_xor(ix, off, 32);
      if (ov > v || (ov == v && oix < ix)) { v = ov; ix = oix; }
    }
    // token M = r + 8*half (C/D 16x16 f32 layout)
    if (m == 0) lds_idx[wave * 16 + half * 8 + r] = ix;
  }
  __syncthreads();

  // ---- output phase: gather codebook rows, write [B,C,L], accumulate loss -
  const int t      = tid & 127;          // token within block (contiguous l)
  const int cpair  = tid >> 7;
  const int tokenO = tok0 + t;
  const int bO = tokenO >> 12;
  const int lO = tokenO & (L_DIM - 1);
  const int myIdx = lds_idx[t];
  const float* crow = cb + (size_t)myIdx * C_DIM;
  float local = 0.0f;
#pragma unroll 4
  for (int i = 0; i < 128; ++i) {
    int c = 2 * i + cpair;
    float e = crow[c];
    size_t off = (size_t)bO * (C_DIM * L_DIM) + (size_t)c * L_DIM + lO;
    float zv = z[off];
    out[off] = e;
    float d = e - zv;
    local += d * d;
  }
  lds_red[tid] = local;
  __syncthreads();
  for (int s = 128; s > 0; s >>= 1) {
    if (tid < s) lds_red[tid] += lds_red[tid + s];
    __syncthreads();
  }
  if (tid == 0) partials[blockIdx.x] = lds_red[0];
}

// ---------------------------------------------------------------------------
// Final deterministic loss reduction: loss = 1.25 * sum / (N*C)
// ---------------------------------------------------------------------------
__global__ void __launch_bounds__(256)
vq_loss(const float* __restrict__ partials, float* __restrict__ out_loss) {
  __shared__ float red[256];
  const int t = threadIdx.x;
  red[t] = partials[t] + partials[t + 256];
  __syncthreads();
  for (int s = 128; s > 0; s >>= 1) {
    if (t < s) red[t] += red[t + s];
    __syncthreads();
  }
  if (t == 0) out_loss[0] = red[0] * (1.25f / 16777216.0f);   // N*C = 2^24
}

extern "C" void kernel_launch(void* const* d_in, const int* in_sizes, int n_in,
                              void* d_out, int out_size, void* d_ws, size_t ws_size,
                              hipStream_t stream) {
  const float* z  = (const float*)d_in[0];   // [16, 256, 4096] f32
  const float* cb = (const float*)d_in[1];   // [1024, 256] f32
  float* out = (float*)d_out;                // z_q [16,256,4096] then loss scalar
  char* ws = (char*)d_ws;

  // workspace layout (~1.03 MB)
  unsigned short* cb_hi = (unsigned short*)ws;                                   // 512 KB
  unsigned short* cb_lo = (unsigned short*)(ws + (size_t)K_CODES * C_DIM * 2);   // 512 KB
  float* halfnorm = (float*)(ws + (size_t)K_CODES * C_DIM * 4);                  // 4 KB
  float* partials = (float*)(ws + (size_t)K_CODES * C_DIM * 4 + K_CODES * 4);    // 2 KB

  vq_prep<<<K_CODES, C_DIM, 0, stream>>>(cb, cb_hi, cb_lo, halfnorm);
  vq_main<<<NBLK, 256, 0, stream>>>(z, cb, cb_hi, cb_lo, halfnorm, out, partials);
  vq_loss<<<1, 256, 0, stream>>>(partials, out + (size_t)N_TOK * C_DIM);
}
